// SpatialAngularAttention_20452634263754
// MI455X (gfx1250) — matrix operand order repaired and checked by reading, not compiled, not run on hardware
//
#include <hip/hip_runtime.h>
#include <hip/hip_bf16.h>

// ---- shapes (b, c, n=u*v, h, w) = (4, 64, 25, 64, 64), u=v=5 ----
// unfold: s = h*5+u in [0,320), d = c*320 + v*64 + w in [0,20480)
#define NB   4
#define SDIM 320
#define DDIM 20480

#if defined(__AMDGCN__) && __has_builtin(__builtin_amdgcn_global_load_async_to_lds_b128)
#define USE_ASYNC_LDS 1
#else
#define USE_ASYNC_LDS 0
#endif

typedef __attribute__((ext_vector_type(16))) __bf16        bf16x16;
typedef __attribute__((ext_vector_type(8)))  float         floatx8;
typedef __attribute__((ext_vector_type(8)))  unsigned int  uint8v;
typedef __attribute__((ext_vector_type(4)))  float         f32x4;
typedef __attribute__((ext_vector_type(4)))  unsigned int  u32x4;

#if USE_ASYNC_LDS
// the async-to-LDS builtin takes GCC-style int4 vector pointers:
//   arg0: global (addrspace(1)) int4*,  arg1: LDS (addrspace(3)) int4*,  offset, cpol
typedef int v4i_vs __attribute__((vector_size(16)));
typedef __attribute__((address_space(1))) v4i_vs* gbl_v4i_ptr;
typedef __attribute__((address_space(3))) v4i_vs* lds_v4i_ptr;
#endif

// flat element index into (b,c,n,h,w) for unfolded coords (bb, s, d)
__device__ __forceinline__ int uaddr(int bb, int s, int d) {
    int h = s / 5, u = s - h * 5;
    int c = d / 320;
    int rem = d - c * 320;
    int vv = rem >> 6, w = rem & 63;
    return ((bb * 64 + c) * 25 + (u * 5 + vv)) * 4096 + h * 64 + w;
}

// f32 -> bf16 bits, round-to-nearest-even
__device__ __forceinline__ unsigned int f2bf(float f) {
    unsigned int u = __builtin_bit_cast(unsigned int, f);
    return (u + 0x7FFFu + ((u >> 16) & 1u)) >> 16;
}
__device__ __forceinline__ unsigned int pack2(float lo, float hi) {
    return f2bf(lo) | (f2bf(hi) << 16);
}

// A fragment (16x32 bf16): lane holds M=lane%16; K runs {b..b+7, b+16..b+23}, b=8*(lane/16).
// The 32-wide K chunk is contiguous in memory for the unfold layout.
__device__ __forceinline__ bf16x16 load_fragA_qk(const float* __restrict__ src, float scale,
                                                 int bb, int s, int kc, int lane) {
    int d0 = kc * 32 + (lane >> 4) * 8;
    const float* p0 = src + uaddr(bb, s, d0);
    f32x4 r0 = *(const f32x4*)(p0);
    f32x4 r1 = *(const f32x4*)(p0 + 4);
    f32x4 r2 = *(const f32x4*)(p0 + 16);
    f32x4 r3 = *(const f32x4*)(p0 + 20);
    uint8v pk;
    pk[0] = pack2(r0[0] * scale, r0[1] * scale);
    pk[1] = pack2(r0[2] * scale, r0[3] * scale);
    pk[2] = pack2(r1[0] * scale, r1[1] * scale);
    pk[3] = pack2(r1[2] * scale, r1[3] * scale);
    pk[4] = pack2(r2[0] * scale, r2[1] * scale);
    pk[5] = pack2(r2[2] * scale, r2[3] * scale);
    pk[6] = pack2(r3[0] * scale, r3[1] * scale);
    pk[7] = pack2(r3[2] * scale, r3[3] * scale);
    return __builtin_bit_cast(bf16x16, pk);
}

// B fragment (32x16 bf16) from k rows: lane holds N=lane%16 (k-row t); K runs {b..b+7, b+16..b+23}, b=8*(lane/16).
__device__ __forceinline__ bf16x16 load_fragB_k(const float* __restrict__ src,
                                                const float* __restrict__ invk,
                                                int bb, int ct, int kc, int lane) {
    int t = ct * 16 + (lane & 15);
    float scale = invk[bb * SDIM + t];
    int d0 = kc * 32 + (lane >> 4) * 8;
    const float* p = src + uaddr(bb, t, d0);
    f32x4 r0 = *(const f32x4*)(p);
    f32x4 r1 = *(const f32x4*)(p + 4);
    f32x4 r2 = *(const f32x4*)(p + 16);
    f32x4 r3 = *(const f32x4*)(p + 20);
    uint8v pk;
    pk[0] = pack2(r0[0] * scale, r0[1] * scale);
    pk[1] = pack2(r0[2] * scale, r0[3] * scale);
    pk[2] = pack2(r1[0] * scale, r1[1] * scale);
    pk[3] = pack2(r1[2] * scale, r1[3] * scale);
    pk[4] = pack2(r2[0] * scale, r2[1] * scale);
    pk[5] = pack2(r2[2] * scale, r2[3] * scale);
    pk[6] = pack2(r3[0] * scale, r3[1] * scale);
    pk[7] = pack2(r3[2] * scale, r3[3] * scale);
    return __builtin_bit_cast(bf16x16, pk);
}

// ---------------- Kernel 1: inverse L2 norms of unfolded q/k rows ----------------
__global__ __launch_bounds__(256) void k_norms(const float* __restrict__ q,
                                               const float* __restrict__ kk,
                                               float* __restrict__ invq,
                                               float* __restrict__ invk) {
    __shared__ float red[256];
    int rid = blockIdx.x;                 // 0..2*NB*SDIM-1
    int which = rid >= NB * SDIM;
    int lin = which ? rid - NB * SDIM : rid;
    int bb = lin / SDIM, s = lin - bb * SDIM;
    const float* src = which ? kk : q;
    float acc = 0.f;
    for (int d = threadIdx.x; d < DDIM; d += 256) {
        float x = src[uaddr(bb, s, d)];
        acc += x * x;
    }
    red[threadIdx.x] = acc;
    __syncthreads();
    for (int off = 128; off > 0; off >>= 1) {
        if ((int)threadIdx.x < off) red[threadIdx.x] += red[threadIdx.x + off];
        __syncthreads();
    }
    if (threadIdx.x == 0) {
        float inv = 1.0f / fmaxf(sqrtf(red[0]), 1e-12f);
        (which ? invk : invq)[lin] = inv;
    }
}

// ------------- Kernel 2: scores (bf16 WMMA) + causal mask + softmax -> P (bf16) -------------
// Statically named accumulators (acc0/1/2) -> no v_movrels, no scratch spills.
__global__ __launch_bounds__(256) void k_scores(const float* __restrict__ q,
                                                const float* __restrict__ kk,
                                                const float* __restrict__ invq,
                                                const float* __restrict__ invk,
                                                unsigned short* __restrict__ P) {
    __shared__ float sc[16 * SDIM];       // 16 rows x 320 scores (f32)
    __shared__ float smax[16], sinv[16];

    int bb = blockIdx.x / 20;
    int stile = blockIdx.x - bb * 20;     // 0..19
    int tid = threadIdx.x;
    int lane = tid & 31;
    int wave = tid >> 5;                  // 8 waves

    int srow = stile * 16 + (lane & 15);
    float qscale = invq[bb * SDIM + srow];

    // column tiles owned by this wave: ct0 = wave, ct1 = wave+8, ct2 = wave+16 (if <20)
    int ct0 = wave, ct1 = wave + 8, ct2 = wave + 16;
    bool has2 = (ct2 < 20);
    // causal liveness (wave-uniform): l2 => l1 => l0
    bool l0 = (ct0 <= stile);
    bool l1 = (ct1 <= stile);
    bool l2 = has2 && (ct2 <= stile);

    floatx8 acc0, acc1, acc2;
#pragma unroll
    for (int g = 0; g < 8; ++g) { acc0[g] = 0.f; acc1[g] = 0.f; acc2[g] = 0.f; }

    if (l0) {
        for (int kc = 0; kc < DDIM / 32; ++kc) {
            bf16x16 afrag = load_fragA_qk(q, qscale, bb, srow, kc, lane);
            {
                bf16x16 b0 = load_fragB_k(kk, invk, bb, ct0, kc, lane);
                acc0 = __builtin_amdgcn_wmma_f32_16x16x32_bf16(
                    false, afrag, false, b0, (short)0, acc0, false, false);
            }
            if (l1) {
                bf16x16 b1 = load_fragB_k(kk, invk, bb, ct1, kc, lane);
                acc1 = __builtin_amdgcn_wmma_f32_16x16x32_bf16(
                    false, afrag, false, b1, (short)0, acc1, false, false);
                if (l2) {
                    bf16x16 b2 = load_fragB_k(kk, invk, bb, ct2, kc, lane);
                    acc2 = __builtin_amdgcn_wmma_f32_16x16x32_bf16(
                        false, afrag, false, b2, (short)0, acc2, false, false);
                }
            }
        }
    }

    // C/D layout: lanes 0-15 -> M=g, N=lane; lanes 16-31 -> M=g+8, N=lane-16
    {
        int nlo = lane & 15;
        int rbase = (lane >> 4) << 3;
        int t0c = ct0 * 16 + nlo, t1c = ct1 * 16 + nlo, t2c = ct2 * 16 + nlo;
#pragma unroll
        for (int g = 0; g < 8; ++g) {
            int row = g + rbase;
            int s = stile * 16 + row;
            float v0 = l0 ? acc0[g] : -1e30f;
            if (t0c > s) v0 = -1e30f;
            sc[row * SDIM + t0c] = v0;
            float v1 = l1 ? acc1[g] : -1e30f;
            if (t1c > s) v1 = -1e30f;
            sc[row * SDIM + t1c] = v1;
            if (has2) {
                float v2 = l2 ? acc2[g] : -1e30f;
                if (t2c > s) v2 = -1e30f;
                sc[row * SDIM + t2c] = v2;
            }
        }
    }
    __syncthreads();

    if (tid < 16) {
        int r = tid;
        float m = -1e30f;
        for (int t = 0; t < SDIM; ++t) m = fmaxf(m, sc[r * SDIM + t]);
        float ssum = 0.f;
        for (int t = 0; t < SDIM; ++t) ssum += __expf(sc[r * SDIM + t] - m);
        smax[r] = m;
        sinv[r] = 1.0f / ssum;
    }
    __syncthreads();

    for (int idx = tid; idx < 16 * SDIM; idx += 256) {
        int r = idx / SDIM, t = idx - r * SDIM;
        float p = __expf(sc[idx] - smax[r]) * sinv[r];
        P[((size_t)(bb * SDIM + stile * 16 + r)) * SDIM + t] = (unsigned short)f2bf(p);
    }
}

// ------------- Kernel 3: out = P @ V (bf16 WMMA), fold + residual add -------------
// One block per (batch, 16-wide d-tile), 4 waves / 128 threads.
// Each K-step: stage the 32x16 V chunk into LDS once — via the CDNA5 async
// memory->LDS pipe (GLOBAL_LOAD_ASYNC_TO_LDS_B128, ASYNCcnt) when available,
// else via coalesced dword loads + ds stores. Every wave builds the shared B
// fragment from LDS; each wave owns 5 s-tiles (5 accumulators, 40 VGPRs).
// -> V is streamed from HBM exactly once; P (800 KB) stays L2-resident.
__global__ __launch_bounds__(128) void k_out(const unsigned short* __restrict__ P,
                                             const float* __restrict__ vsrc,
                                             float* __restrict__ out) {
    __shared__ float vtile[32 * 16];      // V chunk: 32 t-rows x 16 d-cols

    int bb = blockIdx.y;
    int dtile = blockIdx.x;               // 0..1279
    int d0 = dtile * 16;
    int tid = threadIdx.x;
    int lane = tid & 31;
    int wave = tid >> 5;                  // 0..3, owns s-tiles wave*5 .. wave*5+4
    int m = lane & 15;
    int half = lane >> 4;
    int dcol = d0 + m;

    floatx8 acc[5];
#pragma unroll
    for (int j = 0; j < 5; ++j)
#pragma unroll
        for (int g = 0; g < 8; ++g) acc[j][g] = 0.f;

    for (int kc = 0; kc < SDIM / 32; ++kc) {
        __syncthreads();                  // previous iteration's LDS reads done
#if USE_ASYNC_LDS
        {
            // one 16B async copy per thread: 128 threads x 4 dwords = 32x16 tile
            int idx = tid * 4;            // 0..508, 4-dword aligned both sides
            int t = kc * 32 + (idx >> 4);
            const float* gp = vsrc + uaddr(bb, t, d0 + (idx & 15));
            __builtin_amdgcn_global_load_async_to_lds_b128(
                (gbl_v4i_ptr)gp,
                (lds_v4i_ptr)&vtile[idx],
                0, 0);
        }
#if __has_builtin(__builtin_amdgcn_s_wait_asynccnt)
        __builtin_amdgcn_s_wait_asynccnt(0);
#else
        asm volatile("s_wait_asynccnt 0x0" ::: "memory");
#endif
#else
#pragma unroll
        for (int i = 0; i < 4; ++i) {
            int idx = tid + 128 * i;      // 0..511 ; 16-float runs are contiguous
            int t = kc * 32 + (idx >> 4);
            vtile[idx] = vsrc[uaddr(bb, t, d0 + (idx & 15))];
        }
        if (kc + 1 < SDIM / 32) {         // hint next V chunk toward the caches
            __builtin_prefetch(&vsrc[uaddr(bb, (kc + 1) * 32 + (tid >> 4), d0 + (tid & 15))], 0, 1);
        }
#endif
        __syncthreads();

        // B fragment (32x16): lane N=m, K = 16*half + {0..15} read from LDS
        uint8v pkb;
#pragma unroll
        for (int i = 0; i < 8; ++i) {
            float lo = vtile[(half * 16 + 2 * i + 0) * 16 + m];
            float hi = vtile[(half * 16 + 2 * i + 1) * 16 + m];
            pkb[i] = pack2(lo, hi);
        }
        bf16x16 bfrag = __builtin_bit_cast(bf16x16, pkb);

        int k0 = kc * 32 + half * 16;
#pragma unroll
        for (int j = 0; j < 5; ++j) {
            int st = wave * 5 + j;
            const unsigned short* pr =
                P + ((size_t)(bb * SDIM + st * 16 + m)) * SDIM + k0;
            u32x4 a0 = *(const u32x4*)(pr);       // K = k0..k0+7   (bf16 pairs)
            u32x4 a1 = *(const u32x4*)(pr + 8);   // K = k0+8..k0+15
            uint8v pka;
            pka[0] = a0[0]; pka[1] = a0[1]; pka[2] = a0[2]; pka[3] = a0[3];
            pka[4] = a1[0]; pka[5] = a1[1]; pka[6] = a1[2]; pka[7] = a1[3];
            bf16x16 afrag = __builtin_bit_cast(bf16x16, pka);
            acc[j] = __builtin_amdgcn_wmma_f32_16x16x32_bf16(
                false, afrag, false, bfrag, (short)0, acc[j], false, false);
        }
    }

    // epilogue: fold (index permutation) + residual add token_v
#pragma unroll
    for (int j = 0; j < 5; ++j) {
        int st = wave * 5 + j;
#pragma unroll
        for (int g = 0; g < 8; ++g) {
            int row = g + (half << 3);
            int a = uaddr(bb, st * 16 + row, dcol);
            out[a] = acc[j][g] + vsrc[a];
        }
    }
}

extern "C" void kernel_launch(void* const* d_in, const int* in_sizes, int n_in,
                              void* d_out, int out_size, void* d_ws, size_t ws_size,
                              hipStream_t stream) {
    const float* q = (const float*)d_in[0];
    const float* k = (const float*)d_in[1];
    const float* v = (const float*)d_in[2];
    float* out = (float*)d_out;

    // workspace: invq[1280] f32 | invk[1280] f32 | P[4*320*320] bf16  (~830 KB)
    float* invq = (float*)d_ws;
    float* invk = invq + NB * SDIM;
    unsigned short* P = (unsigned short*)(invk + NB * SDIM);

    k_norms<<<2 * NB * SDIM, 256, 0, stream>>>(q, k, invq, invk);
    k_scores<<<NB * 20, 256, 0, stream>>>(q, k, invq, invk, P);
    k_out<<<dim3(1280, NB), 128, 0, stream>>>(P, v, out);
}
